// HybridFIKANLinear_76854144794926
// MI455X (gfx1250) — matrix operand order repaired and checked by reading, not compiled, not run on hardware
//
#include <hip/hip_runtime.h>
#include <hip/hip_bf16.h>
#include <cstdint>

// ---------------------------------------------------------------------------
// HybridFIKAN fused kernel for gfx1250 (MI455X).
//   out = [silu(x) | b_splines(x) | fractal_bases(x)] @ Wpacked  (K=512*16 padded)
// ~64 GFLOP GEMM vs ~58MB HBM traffic -> compute-bound on the matrix pipe.
// Contraction runs on v_wmma_f32_16x16x32_bf16; features are generated on the
// fly into double-buffered LDS so feature VALU co-executes with WMMA (CDNA5
// tracks 16-bit WMMA as TRANS ops -> VALU/XDL co-execution) instead of being
// serialized by bulk-synchronous phases. Packed bf16 weights (4.2MB) stay in L2.
// ---------------------------------------------------------------------------

typedef __attribute__((ext_vector_type(16))) __bf16 v16bf;
typedef __attribute__((ext_vector_type(8)))  float  v8f;

#define BATCH   16384
#define IN      512
#define OUTF    256
#define FEAT    16                 // 15 real features padded to 16 (f=15 -> weight 0)
#define KTOT    (IN * FEAT)        // 8192
#define MT      64                 // rows per workgroup
#define ICHUNK  32                 // inputs per K-chunk
#define KCHUNK  (ICHUNK * FEAT)    // 512 = 16 WMMA k-steps
#define NCHUNKS (IN / ICHUNK)      // 16
#define KSTEPS  (KCHUNK / 32)      // 16
#define WPACK_UINTS (KTOT * OUTF / 2)  // 1,048,576 dwords
#define WPACK_BYTES (KTOT * OUTF * 2)  // 4,194,304 bytes
#define DTAB_ELEMS  (IN * 5)           // 2560 contraction factors
#define ABUF_HALVES (MT * KCHUNK)      // 32768 bf16 per A buffer (64KB)

__device__ __forceinline__ unsigned short f2bf(float f) {
  unsigned int u = __float_as_uint(f);
  u += 0x7FFFu + ((u >> 16) & 1u);            // round-to-nearest-even
  return (unsigned short)(u >> 16);
}

// ---------------------------------------------------------------------------
// Prep: build Wpacked (bf16, WMMA B-fragment swizzled) and d = 0.99*tanh(d_raw).
// B-fragment k order used here (contiguous, same as A; 32x16 B, 8 VGPRs/lane):
//   lanes 0-15 : N = nt*16 + lane,    K = kg*32 + 0..15  (VGPR v -> K=2v,2v+1)
//   lanes 16-31: N = nt*16 + lane-16, K = kg*32 + 16..31
// Fragment fi = kg*16 + nt occupies 1024B; lane's 16 values are 32 contiguous B.
// ---------------------------------------------------------------------------
__global__ __launch_bounds__(256) void fikan_prep(
    const float* __restrict__ bw,   // (O,I)
    const float* __restrict__ sw,   // (O,I,8)
    const float* __restrict__ ss,   // (O,I)
    const float* __restrict__ fw,   // (O,I,6)
    const float* __restrict__ draw, // (I,5)
    unsigned int* __restrict__ wpack,
    float* __restrict__ dtab) {
  int tid = blockIdx.x * 256 + threadIdx.x;
  if (tid < WPACK_UINTS) {
    int fi   = tid >> 8;          // fragment index = kg*16 + nt
    int lane = (tid >> 3) & 31;
    int v    = tid & 7;           // VGPR within fragment
    int kg   = fi >> 4;
    int nt   = fi & 15;
    int n    = nt * 16 + (lane & 15);
    unsigned int pk = 0;
#pragma unroll
    for (int h = 0; h < 2; ++h) {
      int kk = kg * 32 + ((lane >> 4) << 4) + 2 * v + h;   // global K index
      int i  = kk >> 4;
      int f  = kk & 15;
      float val;
      if (f == 0)       val = bw[n * IN + i];
      else if (f < 9)   val = sw[(n * IN + i) * 8 + (f - 1)] * ss[n * IN + i];
      else if (f < 15)  val = fw[(n * IN + i) * 6 + (f - 9)];
      else              val = 0.0f;
      pk |= ((unsigned int)f2bf(val)) << (16 * h);
    }
    wpack[tid] = pk;
  }
  if (tid < DTAB_ELEMS) dtab[tid] = 0.99f * tanhf(draw[tid]);
}

// ---------------------------------------------------------------------------
// Feature generator: 16 features for one (row, input) element.
// ---------------------------------------------------------------------------
__device__ __forceinline__ void compute_features(float xv, int i,
                                                 const float* __restrict__ sD,
                                                 float* __restrict__ feat) {
  // f0: silu
  feat[0] = xv * (1.0f / (1.0f + __expf(-xv)));

  // f1..f8: cubic B-spline bases (de Boor), grid[j] = (j-3)*0.4 - 1
  float b[11];
#pragma unroll
  for (int j = 0; j < 11; ++j) {
    float gj = -1.0f + 0.4f * (float)(j - 3);
    b[j] = (xv >= gj && xv < gj + 0.4f) ? 1.0f : 0.0f;
  }
#pragma unroll
  for (int k = 1; k <= 3; ++k) {
    float inv = 1.0f / (0.4f * (float)k);
#pragma unroll
    for (int j = 0; j + k < 11; ++j) {
      float gj   = -1.0f + 0.4f * (float)(j - 3);
      float gjk1 = -1.0f + 0.4f * (float)(j + k - 2);
      b[j] = (xv - gj) * inv * b[j] + (gjk1 - xv) * inv * b[j + 1];
    }
  }
#pragma unroll
  for (int f = 0; f < 8; ++f) feat[1 + f] = b[f];

  // f9..f14: alpha-fractal hat bases, depth 8 IFS recursion
  float t = fminf(fmaxf(xv, -1.0f), 1.0f);
  float phi[6];
#pragma unroll
  for (int m = 0; m < 6; ++m) {
    float node = -1.0f + 0.4f * (float)m;
    phi[m] = fmaxf(0.0f, 1.0f - fabsf(t - node) * 2.5f);
  }
  float mult = 1.0f;
#pragma unroll
  for (int dth = 0; dth < 8; ++dth) {
    int idx = (int)floorf((t + 1.0f) * 2.5f);
    idx = idx < 0 ? 0 : (idx > 4 ? 4 : idx);
    mult *= sD[i * 5 + idx];
    t = (t - (-1.0f + 0.4f * (float)idx)) * 5.0f - 1.0f;  // (t-t0-idx*h)/h*span+t0
    float u = (t + 1.0f) * 0.5f;
#pragma unroll
    for (int m = 0; m < 6; ++m) {
      float node = -1.0f + 0.4f * (float)m;
      float hatv = fmaxf(0.0f, 1.0f - fabsf(t - node) * 2.5f);
      float base = (m == 0) ? (1.0f - u) : ((m == 5) ? u : 0.0f);
      phi[m] = fmaf(mult, hatv - base, phi[m]);
    }
  }
#pragma unroll
  for (int m = 0; m < 6; ++m) feat[9 + m] = phi[m];
  feat[15] = 0.0f;   // pad
}

// Stage one chunk of features (64 rows x 32 inputs) into an LDS A-buffer.
// NOT unrolled: keeps the hot loop ~1/8 the code size (I$-friendly).
__device__ __forceinline__ void gen_features(
    int c, unsigned short* __restrict__ sAbuf, const float* __restrict__ sD,
    const float* __restrict__ x, int wg, int tid) {
#pragma unroll 1
  for (int pp = 0; pp < (MT * ICHUNK) / 256; ++pp) {
    int pair = pp * 256 + tid;
    int row  = pair >> 5;               // 0..63
    int ic   = pair & 31;               // input within chunk
    int i    = c * ICHUNK + ic;
    float xv = x[(wg * MT + row) * IN + i];
    float feat[FEAT];
    compute_features(xv, i, sD, feat);
    unsigned int pk[8];
#pragma unroll
    for (int q = 0; q < 8; ++q)
      pk[q] = (unsigned int)f2bf(feat[2 * q]) |
              ((unsigned int)f2bf(feat[2 * q + 1]) << 16);
    uint4* dst = (uint4*)&sAbuf[row * KCHUNK + ic * FEAT];
    dst[0] = make_uint4(pk[0], pk[1], pk[2], pk[3]);
    dst[1] = make_uint4(pk[4], pk[5], pk[6], pk[7]);
  }
}

// ---------------------------------------------------------------------------
// Main: 256 WGs x 256 threads (8 waves). WG owns 64 rows x 256 cols.
// Wave (wr, wc): wr=wave>>1 row-tile (16 rows), wc=wave&1 col half (128 cols).
// Software pipeline over K-chunks with double-buffered LDS A: while chunk c is
// consumed by WMMA out of buf[c&1], chunk c+1 features are generated into
// buf[(c+1)&1] inside the SAME barrier interval -> VALU/WMMA overlap.
// ---------------------------------------------------------------------------
__global__ __launch_bounds__(256) void fikan_main(
    const float* __restrict__ x,
    const uint4* __restrict__ wpack,
    const float* __restrict__ dtab,
    float* __restrict__ out) {
  extern __shared__ char smem[];
  unsigned short* sA0 = (unsigned short*)smem;                      // 64KB
  unsigned short* sA1 = (unsigned short*)(smem + ABUF_HALVES * 2);  // 64KB
  float*          sD  = (float*)(smem + 2 * ABUF_HALVES * 2);       // 10KB

  const int tid  = threadIdx.x;
  const int lane = tid & 31;
  const int wave = tid >> 5;
  const int wr   = wave >> 1;       // 0..3 row tile
  const int wc   = wave & 1;        // 0..1 col half
  const int wg   = blockIdx.x;      // 0..255, 64 rows each

  for (int t = tid; t < DTAB_ELEMS; t += 256) sD[t] = dtab[t];
  __syncthreads();                  // sD ready

  v8f acc[8] = {};

  // A-fragment lane addressing (16-bit A 16x32):
  //  lanes 0-15 : M=lane,    K [0..15]
  //  lanes 16-31: M=lane-16, K [16..31]
  const int am = lane & 15;
  const int ak = (lane >> 4) << 4;  // 0 or 16 (in bf16 elements)

  gen_features(0, sA0, sD, x, wg, tid);    // prologue: fill buffer 0
  __syncthreads();

#pragma unroll 1
  for (int c = 0; c < NCHUNKS; ++c) {
    // Produce next chunk into the other buffer (no hazard: its last readers
    // consumed their ds_load results before passing the barrier 1 iter ago).
    if (c + 1 < NCHUNKS)
      gen_features(c + 1, (c + 1) & 1 ? sA1 : sA0, sD, x, wg, tid);

    // Consume current chunk: 16 k-steps x 8 WMMAs per wave.
    const unsigned short* sA = (c & 1) ? sA1 : sA0;
#pragma unroll 1
    for (int ks = 0; ks < KSTEPS; ++ks) {
      const unsigned short* ap = &sA[(wr * 16 + am) * KCHUNK + ks * 32 + ak];
      union { uint4 q[2]; v16bf v; } af;
      af.q[0] = *(const uint4*)(ap);        // K[ak .. ak+7]
      af.q[1] = *(const uint4*)(ap + 8);    // K[ak+8 .. ak+15]
      const int kg = c * KSTEPS + ks;       // global k-step
#pragma unroll
      for (int j = 0; j < 8; ++j) {
        const int nt = wc * 8 + j;
        const uint4* bp = wpack + (unsigned)((kg * 16 + nt) * 64 + lane * 2);
        union { uint4 q[2]; v16bf v; } bf;
        bf.q[0] = bp[0];
        bf.q[1] = bp[1];
        acc[j] = __builtin_amdgcn_wmma_f32_16x16x32_bf16(
            false, af.v, false, bf.v, (short)0, acc[j], false, false);
      }
    }
    __syncthreads();   // next-chunk features complete; safe to flip buffers
  }

  // ---- store: C layout: VGPR r, lanes 0-15 -> M=r, lanes 16-31 -> M=8+r ----
  const int mbase = wg * MT + wr * 16 + ((lane >> 4) << 3);
  const int nlane = lane & 15;
#pragma unroll
  for (int j = 0; j < 8; ++j) {
    const int col = wc * 128 + j * 16 + nlane;
#pragma unroll
    for (int r = 0; r < 8; ++r)
      out[(mbase + r) * OUTF + col] = acc[j][r];
  }
}

extern "C" void kernel_launch(void* const* d_in, const int* in_sizes, int n_in,
                              void* d_out, int out_size, void* d_ws, size_t ws_size,
                              hipStream_t stream) {
  (void)in_sizes; (void)n_in; (void)out_size; (void)ws_size;
  const float* x    = (const float*)d_in[0];
  const float* bw   = (const float*)d_in[1];
  const float* sw   = (const float*)d_in[2];
  const float* ss   = (const float*)d_in[3];
  const float* fw   = (const float*)d_in[4];
  const float* draw = (const float*)d_in[5];

  unsigned int* wpack = (unsigned int*)d_ws;
  float*        dtab  = (float*)((char*)d_ws + WPACK_BYTES);

  fikan_prep<<<WPACK_UINTS / 256, 256, 0, stream>>>(bw, sw, ss, fw, draw,
                                                    wpack, dtab);

  const size_t lds = (size_t)2 * ABUF_HALVES * 2 + DTAB_ELEMS * 4;  // 141,312 B
  fikan_main<<<BATCH / MT, 256, lds, stream>>>(
      x, (const uint4*)d_ws, dtab, (float*)d_out);
}
